// ImprovedAttention_37331855737309
// MI455X (gfx1250) — matrix operand order repaired and checked by reading, not compiled, not run on hardware
//
#include <hip/hip_runtime.h>
#include <hip/hip_bf16.h>
#include <stdint.h>

// ---------------- problem constants (from reference) ----------------
#define BATCH    2
#define DIMC     256
#define HEADS    8
#define HEAD_DIM 32
#define KEY_DIM  16
#define QKV_DIM  512          // 256 + 2*16*8
#define IMG_H    40
#define IMG_W    40
#define NPIX     1600
#define NUM_KEEP 1280         // int(1600 * 0.8)
#define EPSV     1e-12f
#define MTILES   (NPIX / 16)  // 100
#define KCHUNKS  (DIMC / 32)  // 8

// LDS partition sizes for k_gemm256 (floats)
#define XS_SZ (32 * 64)
#define WS_SZ (128 * 32)

// ---------------- vector types ----------------
typedef __attribute__((ext_vector_type(16))) _Float16     v16h;
typedef __attribute__((ext_vector_type(8)))  float        v8f;
typedef __attribute__((ext_vector_type(2)))  float        v2f;
typedef __attribute__((ext_vector_type(4)))  unsigned int u32x4;
typedef __attribute__((ext_vector_type(8)))  int          i32x8;
typedef __attribute__((ext_vector_type(4)))  int          i32x4;

#if __has_builtin(__builtin_amdgcn_tensor_load_to_lds)
#define HAVE_TDM 1
#else
#define HAVE_TDM 0
#endif
#if __has_builtin(__builtin_amdgcn_tensor_store_from_lds)
#define HAVE_TDM_ST 1
#else
#define HAVE_TDM_ST 0
#endif

// ---------------- WMMA wrappers ----------------
__device__ __forceinline__ v8f wmma_f16(v16h a, v16h b, v8f c) {
  // D(f32 16x16) = A(f16 16x32) * B(f16 32x16) + C
  return __builtin_amdgcn_wmma_f32_16x16x32_f16(false, a, false, b, (short)0, c,
                                                false, false);
}

__device__ __forceinline__ v8f wmma_f32k4(v2f a, v2f b, v8f c) {
#if __has_builtin(__builtin_amdgcn_wmma_f32_16x16x4_f32)
  // D(f32 16x16) = A(f32 16x4) * B(f32 4x16) + C   (exact fp32 path)
  return __builtin_amdgcn_wmma_f32_16x16x4_f32(false, a, false, b, (short)0, c,
                                               false, false);
#else
  (void)a; (void)b;
  return c;
#endif
}

// ---------------- TDM descriptors (CDNA5 ISA) ----------------
__device__ __forceinline__ unsigned lds_off(const void* p) {
  // flat LDS address keeps the LDS offset in the low 32 bits (aperture in high bits)
  return (unsigned)(uintptr_t)p;
}

__device__ __forceinline__ void tdm_desc_2d_f32(u32x4& g0, i32x8& g1,
                                                unsigned lds, const float* gptr,
                                                unsigned tile0, unsigned tile1,
                                                unsigned stride0,
                                                unsigned tdim0, unsigned tdim1) {
  unsigned long long ga = (unsigned long long)(uintptr_t)gptr;
  g0.x = 0x1u;                                   // count=1 valid user descriptor
  g0.y = lds;                                    // lds_addr [63:32]
  g0.z = (unsigned)(ga & 0xFFFFFFFFull);         // global_addr [95:64]
  g0.w = (unsigned)((ga >> 32) & 0x01FFFFFFull)  // global_addr [120:96]
       | (2u << 30);                             // type=2 ("image") [127:126]
  g1[0] = (int)(2u << 16);                       // data_size=2 (4B), wg_mask=0
  g1[1] = (int)((tdim0 & 0xFFFFu) << 16);        // tensor_dim0 [79:48] lo
  g1[2] = (int)(((tdim0 >> 16) & 0xFFFFu) | ((tdim1 & 0xFFFFu) << 16));
  g1[3] = (int)(((tdim1 >> 16) & 0xFFFFu) | ((tile0 & 0xFFFFu) << 16)); // tile_dim0
  g1[4] = (int)(tile1 & 0xFFFFu);                // tile_dim1 (rows), tile_dim2=0
  g1[5] = (int)stride0;                          // tensor_dim0_stride lo32
  g1[6] = 0;
  g1[7] = 0;
}

__device__ __forceinline__ void tdm_load_2d_f32(unsigned lds, const float* gsrc,
                                                unsigned tile0, unsigned tile1,
                                                unsigned stride0,
                                                unsigned tdim0, unsigned tdim1) {
#if HAVE_TDM
  u32x4 g0; i32x8 g1;
  tdm_desc_2d_f32(g0, g1, lds, gsrc, tile0, tile1, stride0, tdim0, tdim1);
  i32x4 gz = {0, 0, 0, 0};
#if __clang_major__ >= 23
  i32x8 gz8 = {0, 0, 0, 0, 0, 0, 0, 0};
  __builtin_amdgcn_tensor_load_to_lds(g0, g1, gz, gz, gz8, 0);
#else
  __builtin_amdgcn_tensor_load_to_lds(g0, g1, gz, gz, 0);
#endif
#else
  (void)lds; (void)gsrc; (void)tile0; (void)tile1;
  (void)stride0; (void)tdim0; (void)tdim1;
#endif
}

__device__ __forceinline__ void tdm_store_2d_f32(unsigned lds, const float* gdst,
                                                 unsigned tile0, unsigned tile1,
                                                 unsigned stride0,
                                                 unsigned tdim0, unsigned tdim1) {
#if HAVE_TDM_ST
  u32x4 g0; i32x8 g1;
  tdm_desc_2d_f32(g0, g1, lds, gdst, tile0, tile1, stride0, tdim0, tdim1);
  i32x4 gz = {0, 0, 0, 0};
#if __clang_major__ >= 23
  i32x8 gz8 = {0, 0, 0, 0, 0, 0, 0, 0};
  __builtin_amdgcn_tensor_store_from_lds(g0, g1, gz, gz, gz8, 0);
#else
  __builtin_amdgcn_tensor_store_from_lds(g0, g1, gz, gz, 0);
#endif
#else
  (void)lds; (void)gdst; (void)tile0; (void)tile1;
  (void)stride0; (void)tdim0; (void)tdim1;
#endif
}

// ---------------- wave32 reductions ----------------
__device__ __forceinline__ float wsum(float v)  { for (int o = 16; o; o >>= 1) v += __shfl_xor(v, o, 32); return v; }
__device__ __forceinline__ float wmaxr(float v) { for (int o = 16; o; o >>= 1) v = fmaxf(v, __shfl_xor(v, o, 32)); return v; }
__device__ __forceinline__ float wminr(float v) { for (int o = 16; o; o >>= 1) v = fminf(v, __shfl_xor(v, o, 32)); return v; }
__device__ __forceinline__ int   wisum(int v)   { for (int o = 16; o; o >>= 1) v += __shfl_xor(v, o, 32); return v; }

// =====================================================================
// 1x1-conv GEMM, K=256: out[b,m,n] = sum_k W[m,k]*in[b,k,n] + bias[m]
// Block: 256 thr = 8 waves; tile M=128, N=64.
// TDM double-buffers both the activation chunk (32x64) and the weight
// tile (128x32) in LDS; output leaves via LDS + tensor_store_from_lds.
// Dynamic LDS: 2*8KB (x) + 2*16KB (w) + 32KB (out) = 80 KB.
// NOTE: buffer pointers are computed with index arithmetic only — an
// initialized array of LDS-derived pointers becomes a constant
// aggregate with an addrspacecast that ld.lld cannot relocate.
// =====================================================================
__global__ void k_gemm256(const float* __restrict__ in, const float* __restrict__ w,
                          const float* __restrict__ bias, float* __restrict__ out,
                          int Mtotal) {
  extern __shared__ float gl[];
  // layout: [ xs0 | xs1 | ws0 | ws1 | outs ]
  float* outs = gl + 2 * XS_SZ + 2 * WS_SZ;         // 128 x 64

  const int lane = threadIdx.x & 31;
  const int wave = threadIdx.x >> 5;
  const int half = lane >> 4;
  const int l15  = lane & 15;
  const int n0 = blockIdx.x * 64;
  const int mB = blockIdx.y * 128;
  const int m0 = mB + wave * 16;
  const int b  = blockIdx.z;

  const float* xb = in + (size_t)b * DIMC * NPIX;
  const float* wb = w + (size_t)mB * DIMC;
  __builtin_prefetch(bias + m0, 0, 0);
  v8f acc[4] = {};

#if HAVE_TDM
  if (wave == 0) {   // prologue: chunk 0 into buffer 0
    tdm_load_2d_f32(lds_off(gl), xb + n0, 64, 32, NPIX, NPIX, DIMC);
    tdm_load_2d_f32(lds_off(gl + 2 * XS_SZ), wb, 32, 128, DIMC, DIMC, 128);
  }
#endif

  for (int c = 0; c < KCHUNKS; ++c) {
    const int cur = c & 1;
    float* xbuf = gl + cur * XS_SZ;
    float* wbuf = gl + 2 * XS_SZ + cur * WS_SZ;
#if HAVE_TDM
    if (wave == 0) __builtin_amdgcn_s_wait_tensorcnt(0);
    __syncthreads();                          // chunk c visible to all waves
    if (wave == 0 && c + 1 < KCHUNKS) {       // prefetch chunk c+1 (double buffer)
      const int kn = (c + 1) * 32;
      tdm_load_2d_f32(lds_off(gl + (cur ^ 1) * XS_SZ),
                      xb + (size_t)kn * NPIX + n0, 64, 32, NPIX, NPIX, DIMC);
      tdm_load_2d_f32(lds_off(gl + 2 * XS_SZ + (cur ^ 1) * WS_SZ),
                      wb + kn, 32, 128, DIMC, DIMC, 128);
    }
#else
    const int kc = c * 32;
    __syncthreads();
    for (int i = threadIdx.x; i < XS_SZ; i += 256)
      xbuf[i] = xb[(size_t)(kc + (i >> 6)) * NPIX + n0 + (i & 63)];
    for (int i = threadIdx.x; i < WS_SZ; i += 256)
      wbuf[i] = wb[(size_t)(i >> 5) * DIMC + kc + (i & 31)];
    __syncthreads();
#endif
    // A fragment: weight tile row (16x32, contiguous k = half*16 + j), from LDS
    const float* Wr = wbuf + (wave * 16 + l15) * 32;
    v16h a;
#pragma unroll
    for (int j = 0; j < 16; ++j) a[j] = (_Float16)Wr[half * 16 + j];

#pragma unroll
    for (int ns = 0; ns < 4; ++ns) {
      v16h bf;                    // B fragment: xbuf[k][ns*16 + n] (32x16)
#pragma unroll
      for (int j = 0; j < 16; ++j)
        bf[j] = (_Float16)xbuf[(half * 16 + j) * 64 + ns * 16 + l15];
      acc[ns] = wmma_f16(a, bf, acc[ns]);
    }
    __syncthreads();              // done reading chunk c; buffer reusable at c+2
  }

#if HAVE_TDM_ST
  // stage the 128x64 block result in LDS, then one DMA store
#pragma unroll
  for (int ns = 0; ns < 4; ++ns)
#pragma unroll
    for (int j = 0; j < 8; ++j)
      outs[(wave * 16 + j + half * 8) * 64 + ns * 16 + l15] =
          acc[ns][j] + bias[m0 + j + half * 8];
  __syncthreads();
  if (wave == 0) {
    tdm_store_2d_f32(lds_off(outs), out + ((size_t)b * Mtotal + mB) * NPIX + n0,
                     64, 128, NPIX, NPIX, (unsigned)Mtotal);
    __builtin_amdgcn_s_wait_tensorcnt(0);
  }
#else
#pragma unroll
  for (int ns = 0; ns < 4; ++ns)
#pragma unroll
    for (int j = 0; j < 8; ++j)
      out[((size_t)b * Mtotal + m0 + j + half * 8) * NPIX + n0 + ns * 16 + l15] =
          acc[ns][j] + bias[m0 + j + half * 8];
#endif
}

// =====================================================================
// normalize q,k rows over N (fold temperature into q); one wave per row
// =====================================================================
__global__ void k_norm(float* __restrict__ qkv, const float* __restrict__ temperature) {
  const int lane = threadIdx.x & 31;
  const int wave = threadIdx.x >> 5;
  const int r  = blockIdx.x * 8 + wave;           // 0 .. 511
  const int b  = r / (HEADS * 32);
  const int h  = (r / 32) % HEADS;
  const int ch = r % 32;                          // 0..15 q, 16..31 k
  float* row = qkv + ((size_t)b * QKV_DIM + h * 64 + ch) * NPIX;
  float ss = 0.f;
  for (int i = lane; i < NPIX; i += 32) { float v = row[i]; ss += v * v; }
  ss = wsum(ss);
  float s = 1.f / fmaxf(sqrtf(ss), EPSV);
  if (ch < KEY_DIM) s *= temperature[h];          // fold temperature into q
  for (int i = lane; i < NPIX; i += 32) row[i] *= s;
}

__global__ void k_zero(float* __restrict__ p, int n) {
  int i = blockIdx.x * blockDim.x + threadIdx.x;
  if (i < n) p[i] = 0.f;
}

// =====================================================================
// Fused attention per (n-tile16, head, batch).  Dynamic LDS (300 KB):
//   S    [16][1600] f32  scores / exp values
//   Kbuf [16][1600] f32  k staged once by one TDM load; reused as f16 probs
//   Vbuf [32][1600] f16  v converted once -> b128-friendly fragments
//  Phase 1: S = q^T k   (V_WMMA_F32_16X16X4_F32, exact fp32)
//  Phase 2: per-row top-k threshold (bisection) + weighted softmax
//  Phase 3: out = v * P^T (f16 WMMA, K=1600 split across 4 wave pairs)
// =====================================================================
__global__ void k_attn(const float* __restrict__ qkv, const float* __restrict__ suppr,
                       float* __restrict__ outacc) {
  extern __shared__ float at[];
  float*     S    = at;                              // 16*1600 f32
  float*     Kbuf = at + 16 * NPIX;                  // 16*1600 f32
  _Float16*  Pf   = (_Float16*)(at + 16 * NPIX);     // phase-2 output (aliases Kbuf)
  _Float16*  Vbuf = (_Float16*)(at + 2 * 16 * NPIX); // 32*1600 f16

  const int lane = threadIdx.x & 31;
  const int wave = threadIdx.x >> 5;
  const int half = lane >> 4;
  const int l15  = lane & 15;
  const int nt = blockIdx.x, h = blockIdx.y, b = blockIdx.z;
  const int n0 = nt * 16;

  const float* qb = qkv + ((size_t)b * QKV_DIM + h * 64) * NPIX;  // q rows 0..15
  const float* kb = qb + (size_t)16 * NPIX;                       // k rows (contig)
  const float* vb = qb + (size_t)32 * NPIX;                       // v rows (contig)

  // stage all of k with a single TDM descriptor (1600 x 16 tile)
#if HAVE_TDM
  if (wave == 0) {
    tdm_load_2d_f32(lds_off(Kbuf), kb, /*tile0=*/NPIX, /*tile1=*/16,
                    /*stride0=*/NPIX, NPIX, 16);
    __builtin_amdgcn_s_wait_tensorcnt(0);
  }
#else
  for (int i = threadIdx.x; i < 16 * NPIX; i += 256) Kbuf[i] = kb[i];
#endif
  // stage v as f16 (rows contiguous in qkv)
  for (int i = threadIdx.x; i < 32 * NPIX; i += 256) Vbuf[i] = (_Float16)vb[i];
  __syncthreads();

  // ---- Phase 1: scores (fp32 WMMA, K=16 as 4x K=4); q frags hoisted ----
  v2f aq[4];
#pragma unroll
  for (int kc = 0; kc < 4; ++kc) {
    aq[kc][0] = qb[(size_t)(kc * 4 + half * 2 + 0) * NPIX + n0 + l15];
    aq[kc][1] = qb[(size_t)(kc * 4 + half * 2 + 1) * NPIX + n0 + l15];
  }
  for (int mt = wave; mt < MTILES; mt += 8) {
    v8f c = {};
#pragma unroll
    for (int kc = 0; kc < 4; ++kc) {
      v2f bq;
      bq[0] = Kbuf[(kc * 4 + half * 2 + 0) * NPIX + mt * 16 + l15];
      bq[1] = Kbuf[(kc * 4 + half * 2 + 1) * NPIX + mt * 16 + l15];
      c = wmma_f32k4(aq[kc], bq, c);
    }
#pragma unroll
    for (int j = 0; j < 8; ++j)
      S[(j + half * 8) * NPIX + mt * 16 + l15] = c[j];
  }
  __syncthreads();

  // ---- Phase 2: k-th-value threshold by bisection + weighted softmax ----
  const float scale = 1.f / (1.f + __expf(-suppr[0]));   // sigmoid(suppression)
  for (int rr = wave; rr < 16; rr += 8) {
    float* Sr = S + (size_t)rr * NPIX;
    float mx = -3.4e38f, mn = 3.4e38f;
    for (int i = lane; i < NPIX; i += 32) { float v = Sr[i]; mx = fmaxf(mx, v); mn = fminf(mn, v); }
    mx = wmaxr(mx); mn = wminr(mn);
    float lo = mn, hi = mx;
#pragma unroll 1
    for (int it = 0; it < 25; ++it) {                    // invariant: count(>=lo) >= 1280
      float mid = 0.5f * (lo + hi);
      int cnt = 0;
      for (int i = lane; i < NPIX; i += 32) cnt += (Sr[i] >= mid);
      cnt = wisum(cnt);
      if (cnt >= NUM_KEEP) lo = mid; else hi = mid;
    }
    const float thr = lo;
    float pmax = -3.4e38f;
    for (int i = lane; i < NPIX; i += 32) {
      float v = Sr[i];
      pmax = fmaxf(pmax, v * ((v >= thr) ? 1.f : scale));
    }
    pmax = wmaxr(pmax);
    float sum = 0.f;
    for (int i = lane; i < NPIX; i += 32) {
      float v = Sr[i];
      float e = __expf(v * ((v >= thr) ? 1.f : scale) - pmax);
      Sr[i] = e; sum += e;
    }
    sum = wsum(sum);
    const float inv = 1.f / sum;
    // write normalized probabilities as f16 (b128-friendly for phase 3)
    for (int i = lane; i < NPIX; i += 32) Pf[rr * NPIX + i] = (_Float16)(Sr[i] * inv);
  }
  __syncthreads();

  // ---- Phase 3: out[d, n] = sum_m v[d,m] * P[n,m]  (f16 WMMA, K=1600) ----
  const int dt = wave & 1;            // d-tile (0: d 0..15, 1: d 16..31)
  const int ks = wave >> 1;           // K-slot 0..3
  v8f c = {};
  for (int mc = ks * 32; mc < NPIX; mc += 128) {
    v16h a, bf;
    const _Float16* Vr = Vbuf + (size_t)(dt * 16 + l15) * NPIX + mc;  // A: v 16x32
#pragma unroll
    for (int j = 0; j < 16; ++j) a[j] = Vr[half * 16 + j];
    const _Float16* Pr = Pf + (size_t)l15 * NPIX + mc;                // B: P^T 32x16
#pragma unroll
    for (int j = 0; j < 16; ++j) bf[j] = Pr[half * 16 + j];
    c = wmma_f16(a, bf, c);
  }
#pragma unroll
  for (int j = 0; j < 8; ++j) {
    int d   = dt * 16 + j + half * 8;
    int col = n0 + l15;
    atomicAdd(&outacc[((size_t)b * DIMC + h * 32 + d) * NPIX + col], c[j]);
  }
}

// =====================================================================
// depthwise 3x3 PE conv on v image, accumulate into outacc
// =====================================================================
__global__ void k_pe(const float* __restrict__ qkv, const float* __restrict__ w_pe,
                     const float* __restrict__ b_pe, float* __restrict__ outacc) {
  int idx = blockIdx.x * blockDim.x + threadIdx.x;
  if (idx >= BATCH * DIMC * NPIX) return;
  const int col = idx % NPIX;
  const int c   = (idx / NPIX) % DIMC;
  const int b   = idx / (NPIX * DIMC);
  const int px = col % IMG_W, py = col / IMG_W;
  const int h = c >> 5, d = c & 31;
  const float* vimg = qkv + ((size_t)b * QKV_DIM + h * 64 + 32 + d) * NPIX;
  const float* wk = w_pe + (size_t)c * 9;
  float s = b_pe[c];
#pragma unroll
  for (int dy = -1; dy <= 1; ++dy)
#pragma unroll
    for (int dx = -1; dx <= 1; ++dx) {
      int yy = py + dy, xx = px + dx;
      if (yy >= 0 && yy < IMG_H && xx >= 0 && xx < IMG_W)
        s += wk[(dy + 1) * 3 + (dx + 1)] * vimg[yy * IMG_W + xx];
    }
  outacc[idx] += s;
}

// =====================================================================
extern "C" void kernel_launch(void* const* d_in, const int* in_sizes, int n_in,
                              void* d_out, int out_size, void* d_ws, size_t ws_size,
                              hipStream_t stream) {
  const float* x      = (const float*)d_in[0];
  const float* w_qkv  = (const float*)d_in[1];
  const float* b_qkv  = (const float*)d_in[2];
  const float* w_proj = (const float*)d_in[3];
  const float* b_proj = (const float*)d_in[4];
  const float* w_pe   = (const float*)d_in[5];
  const float* b_pe   = (const float*)d_in[6];
  const float* temp   = (const float*)d_in[7];
  const float* suppr  = (const float*)d_in[8];
  (void)in_sizes; (void)n_in; (void)out_size; (void)ws_size;

  float* qkv = (float*)d_ws;                                    // B*512*1600 f32
  float* acc = qkv + (size_t)BATCH * QKV_DIM * NPIX;            // B*256*1600 f32
  float* out = (float*)d_out;

  const int smem_gemm = (2 * XS_SZ + 2 * WS_SZ + 128 * 64) * (int)sizeof(float);  // 80 KB
  const int smem_attn = (2 * 16 * NPIX) * (int)sizeof(float)                      // S + Kbuf
                      + 32 * NPIX * (int)sizeof(_Float16);                        // Vbuf -> 300 KB
  (void)hipFuncSetAttribute((const void*)k_gemm256,
                            hipFuncAttributeMaxDynamicSharedMemorySize, smem_gemm);
  (void)hipFuncSetAttribute((const void*)k_attn,
                            hipFuncAttributeMaxDynamicSharedMemorySize, smem_attn);

  // qkv = W_qkv @ x + b   (M=512, K=256, N=1600, per batch)
  k_gemm256<<<dim3(NPIX / 64, QKV_DIM / 128, BATCH), 256, smem_gemm, stream>>>(
      x, w_qkv, b_qkv, qkv, QKV_DIM);
  // normalize q,k rows over N; fold temperature into q
  k_norm<<<(BATCH * HEADS * 32) / 8, 256, 0, stream>>>(qkv, temp);
  // zero accumulator
  const int zn = BATCH * DIMC * NPIX;
  k_zero<<<(zn + 255) / 256, 256, 0, stream>>>(acc, zn);
  // fused scores + top-k threshold + softmax + attn*v
  k_attn<<<dim3(MTILES, HEADS, BATCH), 256, smem_attn, stream>>>(qkv, suppr, acc);
  // positional-encoding depthwise conv
  k_pe<<<(zn + 255) / 256, 256, 0, stream>>>(qkv, w_pe, b_pe, acc);
  // final projection: out = W_proj @ acc + b   (M=256, K=256, N=1600)
  k_gemm256<<<dim3(NPIX / 64, DIMC / 128, BATCH), 256, smem_gemm, stream>>>(
      acc, w_proj, b_proj, out, DIMC);
}
